// StochasticBlockModel_91130616087336
// MI455X (gfx1250) — matrix operand order repaired and checked by reading, not compiled, not run on hardware
//
#include <hip/hip_runtime.h>
#include <hip/hip_bf16.h>
#include <math.h>

// SBM EM on MI455X (gfx1250).
// Heavy op per distinct z: M = A @ onehot(z)  via v_wmma_f32_16x16x32_f16
// (A and onehot are binary -> exact in f16; f32 accum -> exact integers).
// 4 passes over A (1 GB HBM) ~= 43 us floor at 23.3 TB/s; all else is tiny.
// A loads are non-temporal (pure stream) so the 256 KB packed-B operand
// stays cache-resident.

#define SBM_N 8192
#define SBM_K 16
#define SBM_EPS 1e-10f

typedef __attribute__((ext_vector_type(16))) _Float16 v16h;
typedef __attribute__((ext_vector_type(8)))  float    v8f;
typedef __attribute__((ext_vector_type(4)))  float    v4f;

// ---------------------------------------------------------------- init
__global__ void sbm_init(const int* __restrict__ z_in, const float* __restrict__ P_in,
                         int* __restrict__ zA, float* __restrict__ PA) {
    int i = blockIdx.x * 256 + threadIdx.x;       // 8192 threads
    zA[i] = z_in[i];
    if (i < 256) PA[i] = P_in[i];
}

// ------------------------------------------------- pack B = onehot(z) into WMMA layout
// Slot g = ks*32 + lane. Lane L: column n = L&15, halves h -> k = ks*32 + (L>=16?16:0) + h.
__global__ void sbm_bpack(const int* __restrict__ z, _Float16* __restrict__ Bp) {
    int g = blockIdx.x * 256 + threadIdx.x;       // 8192 items (256 K-tiles * 32 lanes)
    int L = g & 31;
    int n = L & 15;
    int kbase = ((g >> 5) << 5) + ((L >> 4) << 4);
    v16h v;
#pragma unroll
    for (int h = 0; h < 16; ++h)
        v[h] = (z[kbase + h] == n) ? (_Float16)1.0f : (_Float16)0.0f;
    *(v16h*)(Bp + (size_t)g * 16) = v;
}

// ------------------------------------------------- heavy GEMM: M[8192,16] = A @ onehot(z)
__global__ __launch_bounds__(128) void sbm_gemm(const float* __restrict__ A,
                                                const _Float16* __restrict__ Bp,
                                                float* __restrict__ M) {
    const int lane = threadIdx.x & 31;
    const int wave = threadIdx.x >> 5;
    const int hi   = lane >> 4;                   // lane group (0: lanes 0-15, 1: 16-31)
    const int mt   = blockIdx.x;                  // 512 row-tiles of 16
    const int row  = mt * 16 + (lane & 15);
    const float* __restrict__ arow = A + (size_t)row * SBM_N;

    v8f acc = {};
    const int ks0 = wave * 64;                    // each wave: 64 K-steps (K-chunk of 2048)
#pragma unroll 2
    for (int ks = ks0; ks < ks0 + 64; ++ks) {
        const int k0 = ks << 5;
        // A fragment (16x32 f16 layout): lane holds row, runs [k0+16*hi..+15] (contiguous, matches Bpack).
        // Non-temporal: A is streamed once per pass; don't let it thrash L2 (Bpack lives there).
        const v4f* pa = (const v4f*)(arow + k0 + hi * 16);
        const v4f* pb = (const v4f*)(arow + k0 + hi * 16 + 8);
        v4f f0 = __builtin_nontemporal_load(pa);
        v4f f1 = __builtin_nontemporal_load(pa + 1);
        v4f f2 = __builtin_nontemporal_load(pb);
        v4f f3 = __builtin_nontemporal_load(pb + 1);
        v16h av;
#pragma unroll
        for (int e = 0; e < 4; ++e) {
            av[e]      = (_Float16)f0[e];
            av[4 + e]  = (_Float16)f1[e];
            av[8 + e]  = (_Float16)f2[e];
            av[12 + e] = (_Float16)f3[e];
        }
        // B fragment: prepacked, contiguous 32B per (ks,lane); L2-resident
        v16h bv = *(const v16h*)(Bp + (((size_t)ks << 5) + lane) * 16);
        acc = __builtin_amdgcn_wmma_f32_16x16x32_f16(false, av, false, bv,
                                                     (short)0, acc, false, false);
    }

    // reduce 4 wave-partials through LDS; wave 0 stores tile
    __shared__ float red[4][32][8];
#pragma unroll
    for (int v = 0; v < 8; ++v) red[wave][lane][v] = acc[v];
    __syncthreads();
    if (wave == 0) {
#pragma unroll
        for (int v = 0; v < 8; ++v) {
            float s = red[0][lane][v] + red[1][lane][v] + red[2][lane][v] + red[3][lane][v];
            int m = v + hi * 8;                               // C/D layout: VGPR v, hi-half -> +8
            __builtin_nontemporal_store(
                s, M + (size_t)(mt * 16 + m) * SBM_K + (lane & 15));
        }
    }
}

// ------------------------------------------------- tables: Lp-Lq, Lq, Sq[k]=sum_j Lq[k,z_j]
__global__ void sbm_prep(const float* __restrict__ P, const int* __restrict__ z,
                         float* __restrict__ lpmq, float* __restrict__ lq,
                         float* __restrict__ sq) {
    __shared__ int   hist[16];
    __shared__ float slq[256];
    int t = threadIdx.x;                          // 256 threads
    if (t < 16) hist[t] = 0;
    __syncthreads();
    for (int j = t; j < SBM_N; j += 256) atomicAdd(&hist[z[j]], 1);
    float p   = P[t];
    float vlq = logf(1.0f - p + SBM_EPS);
    float vlp = logf(p + SBM_EPS);
    slq[t] = vlq;
    lpmq[t] = vlp - vlq;
    lq[t]   = vlq;
    __syncthreads();
    if (t < 16) {
        float s = 0.0f;
#pragma unroll
        for (int b = 0; b < 16; ++b) s += (float)hist[b] * slq[t * 16 + b];
        sq[t] = s;
    }
}

// ------------------------------------------------- E-step: logits -> resp (softmax) + argmax
__global__ void sbm_estep(const float* __restrict__ M, const float* __restrict__ lpmq,
                          const float* __restrict__ lq, const float* __restrict__ sq,
                          const int* __restrict__ z, float* __restrict__ resp,
                          int* __restrict__ znext) {
    __shared__ float sLpmq[256], sLq[256], sSq[16];
    int t = threadIdx.x;
    sLpmq[t] = lpmq[t];
    sLq[t]   = lq[t];
    if (t < 16) sSq[t] = sq[t];
    __syncthreads();

    int i  = blockIdx.x * 256 + t;
    int zi = z[i];
    const v4f* mp = (const v4f*)(M + (size_t)i * SBM_K);
    v4f m0 = mp[0], m1 = mp[1], m2 = mp[2], m3 = mp[3];
    float mr[16];
#pragma unroll
    for (int e = 0; e < 4; ++e) {
        mr[e] = m0[e]; mr[4 + e] = m1[e]; mr[8 + e] = m2[e]; mr[12 + e] = m3[e];
    }

    float logit[16];
    float mx = -1e30f;
    int   bi = 0; float best = -1e30f;
#pragma unroll
    for (int k = 0; k < 16; ++k) {
        float s = sSq[k] - sLq[k * 16 + zi];
#pragma unroll
        for (int b = 0; b < 16; ++b) s += mr[b] * sLpmq[k * 16 + b];
        logit[k] = s;
        mx = fmaxf(mx, s);
        if (s > best) { best = s; bi = k; }       // first-occurrence ties like jnp.argmax
    }
    float den = 0.0f;
#pragma unroll
    for (int k = 0; k < 16; ++k) { float e = expf(logit[k] - mx); logit[k] = e; den += e; }
    float inv = 1.0f / den;
#pragma unroll
    for (int k = 0; k < 16; ++k)
        __builtin_nontemporal_store(logit[k] * inv, resp + (size_t)i * SBM_K + k);
    znext[i] = bi;
}

// ------------------------------------------------- zero K*K tallies
__global__ void sbm_zero(float* __restrict__ edge, float* __restrict__ nvec) {
    int t = threadIdx.x;                          // 320 threads
    if (t < 256) edge[t] = 0.0f;
    else if (t < 272) nvec[t - 256] = 0.0f;
}

// ------------------------------------------------- edge[a,b] = sum_i [z_i=a] M[i,b]; counts
__global__ void sbm_edge(const float* __restrict__ M, const int* __restrict__ z,
                         float* __restrict__ edge, float* __restrict__ nvec) {
    __shared__ float sE[256];
    __shared__ int   sH[16];
    int t = threadIdx.x;
    sE[t] = 0.0f;
    if (t < 16) sH[t] = 0;
    __syncthreads();
    int i = blockIdx.x * 256 + t;
    int a = z[i];
    atomicAdd(&sH[a], 1);
    const float* mi = M + (size_t)i * SBM_K;
#pragma unroll
    for (int b = 0; b < 16; ++b) atomicAdd(&sE[a * 16 + b], mi[b]);
    __syncthreads();
    atomicAdd(&edge[t], sE[t]);
    if (t < 16) atomicAdd(&nvec[t], (float)sH[t]);
}

// ------------------------------------------------- P = edge / max(cnt, 1)
__global__ void sbm_pstep(const float* __restrict__ edge, const float* __restrict__ nvec,
                          float* __restrict__ Pout) {
    int t = threadIdx.x;                          // 256 threads
    int a = t >> 4, b = t & 15;
    float na = nvec[a], nb = nvec[b];
    float cnt = na * nb - (a == b ? na : 0.0f);
    Pout[t] = edge[t] / fmaxf(cnt, 1.0f);
}

// ------------------------------------------------- write P, z(float), ll to d_out
__global__ void sbm_final(const float* __restrict__ P, const int* __restrict__ z,
                          const float* __restrict__ edge, const float* __restrict__ nvec,
                          float* __restrict__ out) {
    int t = threadIdx.x;
    int i = blockIdx.x * 256 + t;
    out[SBM_N * SBM_K + 256 + i] = (float)z[i];   // z as float
    if (blockIdx.x != 0) return;
    float p = P[t];
    out[SBM_N * SBM_K + t] = p;                   // P
    int a = t >> 4, b = t & 15;
    float na = nvec[a], nb = nvec[b];
    float cnt  = na * nb - (a == b ? na : 0.0f);
    float term = edge[t] * logf(p + SBM_EPS) + (cnt - edge[t]) * logf(1.0f - p + SBM_EPS);
    __shared__ float sred[256];
    sred[t] = term;
    __syncthreads();
    for (int s = 128; s > 0; s >>= 1) {
        if (t < s) sred[t] += sred[t + s];
        __syncthreads();
    }
    if (t == 0) out[SBM_N * SBM_K + 256 + SBM_N] = 0.5f * sred[0];   // ll
}

// ---------------------------------------------------------------- launcher
extern "C" void kernel_launch(void* const* d_in, const int* in_sizes, int n_in,
                              void* d_out, int out_size, void* d_ws, size_t ws_size,
                              hipStream_t stream) {
    (void)in_sizes; (void)n_in; (void)out_size; (void)ws_size;
    const float* A    = (const float*)d_in[0];    // [8192,8192]
    const float* P_in = (const float*)d_in[1];    // [16,16]
    const int*   z_in = (const int*)d_in[2];      // [8192]
    float* out = (float*)d_out;

    // workspace carve-up (~857 KB total)
    char* w = (char*)d_ws;
    _Float16* Bpack = (_Float16*)w;                               // 256 KB
    float*    Mbuf  = (float*)(w + 262144);                       // 512 KB
    int*      zA    = (int*)(w + 262144 + 524288);                // 32 KB
    int*      zB    = zA + SBM_N;                                 // 32 KB
    float*    PA    = (float*)(zB + SBM_N);
    float*    PB    = PA + 256;
    float*    lpmq  = PB + 256;
    float*    lqt   = lpmq + 256;
    float*    sqt   = lqt + 256;
    float*    edge  = sqt + 16;
    float*    nvec  = edge + 256;

    sbm_init<<<32, 256, 0, stream>>>(z_in, P_in, zA, PA);
    sbm_bpack<<<32, 256, 0, stream>>>(zA, Bpack);
    sbm_gemm<<<512, 128, 0, stream>>>(A, Bpack, Mbuf);            // M = A @ onehot(z0)

    int*   zc = zA; int*   zn = zB;
    float* Pc = PA; float* Pn = PB;
    for (int it = 0; it < 3; ++it) {                              // n_iter == 3 (fixed)
        sbm_prep<<<1, 256, 0, stream>>>(Pc, zc, lpmq, lqt, sqt);
        sbm_estep<<<32, 256, 0, stream>>>(Mbuf, lpmq, lqt, sqt, zc, out, zn);
        sbm_bpack<<<32, 256, 0, stream>>>(zn, Bpack);
        sbm_gemm<<<512, 128, 0, stream>>>(A, Bpack, Mbuf);        // M = A @ onehot(z_new)
        sbm_zero<<<1, 320, 0, stream>>>(edge, nvec);
        sbm_edge<<<32, 256, 0, stream>>>(Mbuf, zn, edge, nvec);
        sbm_pstep<<<1, 256, 0, stream>>>(edge, nvec, Pn);
        int* tz = zc; zc = zn; zn = tz;
        float* tp = Pc; Pc = Pn; Pn = tp;
    }
    sbm_final<<<32, 256, 0, stream>>>(Pc, zc, edge, nvec, out);
}
